// NNMod_31267361915408
// MI455X (gfx1250) — matrix operand order repaired and checked by reading, not compiled, not run on hardware
//
#include <hip/hip_runtime.h>

#define T_ 4
#define N_ 65536
#define D_ 256
#define EPS_ 1e-5f
#define TILE_M 64
#define LDH 264        // bf16 elems per activation LDS row (256 + 8 pad -> 528B stride)
#define LDF 260        // f32 elems per LDS row (256 + 4 pad)
#define CHUNK_LDK 40   // bf16 elems per weight-chunk LDS row (32 + 8 pad -> 80B stride)
#define CHUNK_ELEMS (D_ * CHUNK_LDK)   // one K=32 chunk for all 256 n

typedef __bf16 bf16_t;
typedef __attribute__((ext_vector_type(16))) __bf16 v16bf;
typedef __attribute__((ext_vector_type(8)))  __bf16 v8bf;
typedef __attribute__((ext_vector_type(8)))  float  v8f;

#if __has_builtin(__builtin_amdgcn_global_load_async_to_lds_b128)
#define HAVE_ASYNC_LDS 1
#else
#define HAVE_ASYNC_LDS 0
#endif

// exact param types per hipcc diagnostic: int4 (gcc vector_size) pointers,
// global source in AS(1) (printed as __device__), LDS dest in AS(3)
typedef int v4i_ __attribute__((vector_size(16)));
typedef v4i_ __attribute__((address_space(1))) gv4i;
typedef v4i_ __attribute__((address_space(3))) lv4i;

// ---- 16B global -> LDS copy: async (ASYNCcnt) if available, else via VGPRs
__device__ __forceinline__ void wg_copy16(bf16_t* lds_dst, const bf16_t* gsrc) {
#if HAVE_ASYNC_LDS
    __builtin_amdgcn_global_load_async_to_lds_b128(
        (gv4i*)(uintptr_t)gsrc,
        (lv4i*)(uint32_t)(uintptr_t)lds_dst,
        0, 0);
#else
    uint4 v = *(const uint4*)gsrc;
    *(uint4*)lds_dst = v;
#endif
}

template <int N>
__device__ __forceinline__ void wait_async() {
#if HAVE_ASYNC_LDS
#if __has_builtin(__builtin_amdgcn_s_wait_asynccnt)
    __builtin_amdgcn_s_wait_asynccnt(N);
#else
    asm volatile("s_wait_asynccnt %0" :: "i"(N));
#endif
#endif
}

// Stage one K=32 chunk (all 256 rows of W[n][k]) into LDS. 256 threads: one row each,
// 64B = 4 x b128 per thread.
__device__ __forceinline__ void copy_chunk(bf16_t* dst, const bf16_t* Wt, int c, int tid) {
    const bf16_t* src = Wt + (size_t)tid * D_ + c * 32;
    bf16_t* d = dst + tid * CHUNK_LDK;
#pragma unroll
    for (int j = 0; j < 4; ++j) wg_copy16(d + j * 8, src + j * 8);
}

// ---------------------------------------------------------------- bucketing
__global__ void zero_counts_kernel(int* counts) {
    if (threadIdx.x < T_) counts[threadIdx.x] = 0;
}

__global__ void bucket_kernel(const int* __restrict__ species,
                              int* __restrict__ counts,
                              int* __restrict__ idx) {
    int i = blockIdx.x * blockDim.x + threadIdx.x;   // grid covers N_ exactly
    int s = species[i];
    int p = atomicAdd(&counts[s], 1);
    idx[s * N_ + p] = i;
}

// ------------------------------------------------- f32 -> bf16 weight stage
__global__ void convert_kernel(const float* __restrict__ W0,
                               const float* __restrict__ Wa,
                               const float* __restrict__ Wl,
                               const float* __restrict__ Wf,
                               bf16_t* __restrict__ out) {
    const int M = T_ * D_ * D_;
    int i = blockIdx.x * blockDim.x + threadIdx.x;   // grid covers M exactly
    out[0 * M + i] = (bf16_t)W0[i];
    out[1 * M + i] = (bf16_t)Wa[i];
    out[2 * M + i] = (bf16_t)Wl[i];
    out[3 * M + i] = (bf16_t)Wf[i];
}

// --------------------------------------------------------- WMMA fragments
// A 16x32 bf16 (M x K): lanes 0-15 = rows, K {0..7,16..23}; lanes 16-31: K {8..15,24..31}
__device__ __forceinline__ v16bf load_a_frag(const bf16_t* X, int row0, int k0, int lane) {
    int m  = lane & 15;
    int kh = (lane >> 4) & 1;
    const bf16_t* p = X + (row0 + m) * LDH + k0 + kh * 8;
    union { v16bf v; v8bf h[2]; } u;
    u.h[0] = *(const v8bf*)(p);
    u.h[1] = *(const v8bf*)(p + 16);
    return u.v;
}

// B 32x16 bf16 (K x N) from the LDS chunk: lane holds column n0+(lane&15);
// lanes 0-15: chunk K {0..7,16..23}, lanes 16-31: chunk K {8..15,24..31}.
__device__ __forceinline__ v16bf load_b_frag_lds(const bf16_t* Wc, int n0, int lane) {
    int n  = n0 + (lane & 15);
    int kh = (lane >> 4) & 1;
    const bf16_t* p = Wc + n * CHUNK_LDK + kh * 8;
    union { v16bf v; v8bf h[2]; } u;
    u.h[0] = *(const v8bf*)(p);
    u.h[1] = *(const v8bf*)(p + 16);
    return u.v;
}

// One [64 x 256] = A[64x256] @ W^T stage. K-outer with double-buffered LDS weight
// chunks: async-copy chunk c+1 while the 8 WMMAs of chunk c run.
// wave w: mt = w>>1 (16 rows), nh = w&1 (128 cols as 8 tiles of 16).
template <typename F>
__device__ __forceinline__ void gemm_stage_lds(const bf16_t* XhIn, const bf16_t* Wt,
                                               bf16_t* Wc, int tid, int lane,
                                               int mt, int nh, F&& post) {
    v8f acc[8];
#pragma unroll
    for (int nt = 0; nt < 8; ++nt)
        acc[nt] = v8f{0.f, 0.f, 0.f, 0.f, 0.f, 0.f, 0.f, 0.f};

    copy_chunk(Wc, Wt, 0, tid);
#pragma unroll
    for (int c = 0; c < 8; ++c) {
        if (c < 7) {
            copy_chunk(Wc + ((c + 1) & 1) * CHUNK_ELEMS, Wt, c + 1, tid);
            wait_async<4>();          // chunk c landed (async dones are in-order)
        } else {
            wait_async<0>();
        }
        __syncthreads();              // chunk c visible to all waves
        v16bf a = load_a_frag(XhIn, mt * 16, c * 32, lane);
        const bf16_t* Wbp = Wc + (c & 1) * CHUNK_ELEMS;
#pragma unroll
        for (int nt = 0; nt < 8; ++nt) {
            v16bf b = load_b_frag_lds(Wbp, nh * 128 + nt * 16, lane);
            acc[nt] = __builtin_amdgcn_wmma_f32_16x16x32_bf16(false, a, false, b,
                                                              (short)0, acc[nt],
                                                              false, false);
        }
        __syncthreads();              // everyone done reading buf (c&1) before reuse
    }
    // C/D layout: lane = column n0+(lane&15); VGPR r = row; lanes>=16 rows +8
    const int mbase = mt * 16 + ((lane >> 4) & 1) * 8;
#pragma unroll
    for (int nt = 0; nt < 8; ++nt) {
        const int n = nh * 128 + nt * 16 + (lane & 15);
#pragma unroll
        for (int r = 0; r < 8; ++r) post(nt, mbase + r, n, acc[nt][r]);
    }
}

__device__ __forceinline__ float silu(float v) {
    return v / (1.f + __expf(-v));
}

// --------------------------------------------------------- fused expert
__global__ void __launch_bounds__(256)
expert_kernel(const float* __restrict__ density,
              const int*   __restrict__ idx,
              const int*   __restrict__ counts,
              const bf16_t* __restrict__ Wb,
              const float* __restrict__ b0,
              const float* __restrict__ ba,
              const float* __restrict__ alpha_a,
              const float* __restrict__ gam,
              const float* __restrict__ beta_ln,
              const float* __restrict__ bl,
              const float* __restrict__ bfv,
              const float* __restrict__ alpha_f,
              const float* __restrict__ Wo,
              const float* __restrict__ bo,
              float* __restrict__ out)
{
    __shared__ bf16_t XhA[TILE_M * LDH];       // ping (bf16 activations)
    __shared__ bf16_t XhB[TILE_M * LDH];       // pong
    __shared__ float  Xf [TILE_M * LDF];       // f32 side buffer (x1, then x3)
    __shared__ bf16_t Wc [2 * CHUNK_ELEMS];    // double-buffered weight K-chunks

    const int t    = blockIdx.x >> 10;         // 1024 tiles per species
    const int tile = blockIdx.x & 1023;
    const int cnt  = counts[t];
    const int m0   = tile * TILE_M;
    if (m0 >= cnt) return;

    const int tid  = threadIdx.x;
    const int lane = tid & 31;
    const int w    = tid >> 5;                 // 8 waves
    const int mt   = w >> 1;
    const int nh   = w & 1;

    const int MAT = T_ * D_ * D_;
    const bf16_t* W0t = Wb + 0 * MAT + t * D_ * D_;
    const bf16_t* Wat = Wb + 1 * MAT + t * D_ * D_;
    const bf16_t* Wlt = Wb + 2 * MAT + t * D_ * D_;
    const bf16_t* Wft = Wb + 3 * MAT + t * D_ * D_;
    const float* b0t = b0      + t * D_;
    const float* bat = ba      + t * D_;
    const float* aat = alpha_a + t * D_;
    const float* gmt = gam     + t * D_;
    const float* btt = beta_ln + t * D_;
    const float* blt = bl      + t * D_;
    const float* bft = bfv     + t * D_;
    const float* aft = alpha_f + t * D_;

    // per-lane column indices for this wave's 8 N-tiles (for param preloads)
    const int ncol = nh * 128 + (lane & 15);

    // ---- gather density rows for this species tile -> XhA (bf16)
    for (int i = tid; i < TILE_M * 64; i += 256) {   // 64 float4 per row
        int row = i >> 6;
        int c4  = i & 63;
        float4 v = make_float4(0.f, 0.f, 0.f, 0.f);
        int g = m0 + row;
        if (g < cnt) {
            int atom = idx[t * N_ + g];
            v = ((const float4*)density)[(size_t)atom * 64 + c4];
        }
        bf16_t* p = &XhA[row * LDH + c4 * 4];
        p[0] = (bf16_t)v.x; p[1] = (bf16_t)v.y; p[2] = (bf16_t)v.z; p[3] = (bf16_t)v.w;
    }
    __syncthreads();

    // ---- stage 0: x1 = density @ W0^T + b0  (f32 for residual, bf16 as next A)
    {
        float p0[8];
#pragma unroll
        for (int nt = 0; nt < 8; ++nt) p0[nt] = b0t[ncol + nt * 16];
        gemm_stage_lds(XhA, W0t, Wc, tid, lane, mt, nh,
                       [&](int nt, int m, int n, float v) {
            v += p0[nt];
            Xf [m * LDF + n] = v;
            XhB[m * LDH + n] = (bf16_t)v;
        });
    }
    __syncthreads();

    // ---- stage 1: h = alpha_a * silu(x1 @ Wa^T + ba)
    {
        float p0[8], p1[8];
#pragma unroll
        for (int nt = 0; nt < 8; ++nt) {
            p0[nt] = bat[ncol + nt * 16];
            p1[nt] = aat[ncol + nt * 16];
        }
        gemm_stage_lds(XhB, Wat, Wc, tid, lane, mt, nh,
                       [&](int nt, int m, int n, float v) {
            XhA[m * LDH + n] = (bf16_t)(p1[nt] * silu(v + p0[nt]));
        });
    }
    __syncthreads();

    // ---- LayerNorm over D, 8 rows per wave, wave32 shfl reductions
    for (int rr = 0; rr < 8; ++rr) {
        int m = w * 8 + rr;
        float vals[8];
        float s = 0.f, s2 = 0.f;
#pragma unroll
        for (int j = 0; j < 8; ++j) {
            float h = (float)XhA[m * LDH + lane * 8 + j];
            vals[j] = h; s += h; s2 += h * h;
        }
#pragma unroll
        for (int off = 16; off >= 1; off >>= 1) {
            s  += __shfl_xor(s,  off, 32);
            s2 += __shfl_xor(s2, off, 32);
        }
        float mu  = s * (1.f / 256.f);
        float var = s2 * (1.f / 256.f) - mu * mu;
        float inv = rsqrtf(var + EPS_);
#pragma unroll
        for (int j = 0; j < 8; ++j) {
            int n = lane * 8 + j;
            XhA[m * LDH + n] = (bf16_t)(gmt[n] * (vals[j] - mu) * inv + btt[n]);
        }
    }
    __syncthreads();

    // ---- stage 2: x2 = x1 + h_ln @ Wl^T + bl
    {
        float p0[8];
#pragma unroll
        for (int nt = 0; nt < 8; ++nt) p0[nt] = blt[ncol + nt * 16];
        gemm_stage_lds(XhA, Wlt, Wc, tid, lane, mt, nh,
                       [&](int nt, int m, int n, float v) {
            XhB[m * LDH + n] = (bf16_t)(Xf[m * LDF + n] + v + p0[nt]);
        });
    }
    __syncthreads();

    // ---- stage 3: x3 = alpha_f * silu(x2 @ Wf^T + bf)  -> f32 buffer
    {
        float p0[8], p1[8];
#pragma unroll
        for (int nt = 0; nt < 8; ++nt) {
            p0[nt] = bft[ncol + nt * 16];
            p1[nt] = aft[ncol + nt * 16];
        }
        gemm_stage_lds(XhB, Wft, Wc, tid, lane, mt, nh,
                       [&](int nt, int m, int n, float v) {
            Xf[m * LDF + n] = p1[nt] * silu(v + p0[nt]);
        });
    }
    __syncthreads();

    // ---- head: y = x3 . Wo[t] + bo[t]
    const float* wo  = Wo + t * D_;
    const float  bov = bo[t];
    for (int rr = 0; rr < 8; ++rr) {
        int m = w * 8 + rr;
        float s = 0.f;
#pragma unroll
        for (int j = 0; j < 8; ++j)
            s += Xf[m * LDF + lane * 8 + j] * wo[lane * 8 + j];
#pragma unroll
        for (int off = 16; off >= 1; off >>= 1) s += __shfl_xor(s, off, 32);
        if (lane == 0) {
            int g = m0 + m;
            if (g < cnt) out[idx[t * N_ + g]] = s + bov;
        }
    }
}

// ---------------------------------------------------------------- launcher
extern "C" void kernel_launch(void* const* d_in, const int* in_sizes, int n_in,
                              void* d_out, int out_size, void* d_ws, size_t ws_size,
                              hipStream_t stream) {
    (void)in_sizes; (void)n_in; (void)out_size; (void)ws_size;

    const float* density = (const float*)d_in[0];
    const int*   species = (const int*)  d_in[1];
    const float* W0      = (const float*)d_in[2];
    const float* b0      = (const float*)d_in[3];
    const float* Wa      = (const float*)d_in[4];
    const float* ba      = (const float*)d_in[5];
    const float* alpha_a = (const float*)d_in[6];
    const float* gam     = (const float*)d_in[7];
    const float* beta_ln = (const float*)d_in[8];
    const float* Wl      = (const float*)d_in[9];
    const float* bl      = (const float*)d_in[10];
    const float* Wf      = (const float*)d_in[11];
    const float* bfv     = (const float*)d_in[12];
    const float* alpha_f = (const float*)d_in[13];
    const float* Wo      = (const float*)d_in[14];
    const float* bo      = (const float*)d_in[15];
    float* out = (float*)d_out;

    char* ws = (char*)d_ws;
    int*    counts = (int*)ws;                                   // 4 ints (pad to 256B)
    int*    idx    = (int*)(ws + 256);                           // T_*N_ ints = 1 MB
    bf16_t* Wb     = (bf16_t*)(ws + 256 + (size_t)T_ * N_ * 4);  // 4*T_*D_*D_ bf16 = 2 MB

    zero_counts_kernel<<<1, 64, 0, stream>>>(counts);
    bucket_kernel<<<N_ / 256, 256, 0, stream>>>(species, counts, idx);
    convert_kernel<<<(T_ * D_ * D_) / 256, 256, 0, stream>>>(W0, Wa, Wl, Wf, Wb);
    expert_kernel<<<T_ * (N_ / TILE_M), 256, 0, stream>>>(
        density, idx, counts, Wb,
        b0, ba, alpha_a, gam, beta_ln, bl, bfv, alpha_f, Wo, bo, out);
}
